// ANIMAZeroExact_86887188398421
// MI455X (gfx1250) — matrix operand order repaired and checked by reading, not compiled, not run on hardware
//
#include <hip/hip_runtime.h>

// ---------------------------------------------------------------------------
// ANIMA recurrent scan for MI455X (gfx1250, wave32, WMMA).
//
// Roofline: ~83 GFLOP of 16x16xK GEMMs, <30 MB HBM traffic; weights live in
// LDS (320KB/WGP). Sequential in T => per-wave batch tile of 16 rows, all
// matmuls (incl. the padded encoder and the padded N=4 phi head) as
// V_WMMA_F32_16X16X32_F16 chains over LDS-resident f16 state. Activations use
// native v_tanh_f32 / v_exp_f32 / v_rcp_f32 instead of IEEE divides.
// ---------------------------------------------------------------------------

#define BB 1024
#define TT 512
#define SS 8
#define OO 4
#define DD 64

typedef _Float16 h16 __attribute__((ext_vector_type(16)));
typedef _Float16 h8  __attribute__((ext_vector_type(8)));
typedef float    f8  __attribute__((ext_vector_type(8)));

// ---- LDS layout (units: _Float16 elements) --------------------------------
// Weights transposed to [n][k] with padded pitch (pitch = K + 16, keeps every
// row 32B aligned so B-fragments are two ds_load_b128 per lane).
#define ENC_P   48              // K = 32 (padded from 8)
#define KP192   208             // K = 192
#define KP128   144             // K = 128
#define PHI_P   80              // K = 64, N padded 4 -> 16
#define ENC_OFF 0
#define WIN_OFF (ENC_OFF + 64*ENC_P)     // stacked [W_from_W; W_from_I; W_from_A]
#define WG_OFF  (WIN_OFF + 64*KP192)
#define IZ_OFF  (WG_OFF  + 64*KP128)
#define IR_OFF  (IZ_OFF  + 64*KP192)
#define IH_OFF  (IR_OFF  + 64*KP192)
#define AIN_OFF (IH_OFF  + 64*KP192)     // stacked [A_from_W; A_from_I; A_from_A]
#define AG_OFF  (AIN_OFF + 64*KP192)
#define PHI_OFF (AG_OFF  + 64*KP128)
#define ST_OFF  (PHI_OFF + 16*PHI_P)
// Per-wave state strip: 16 rows x 256 cols (+pad): cols [0:64)=W  [64:128)=I
// [128:192)=A  [192:256)=r*I scratch.
#define ST_P    272
#define ST_WAVE (16*ST_P)
#define FLT_OFF (ST_OFF + 4*ST_WAVE)     // halves; float region starts here
#define NFLOATS (64*6 + 16)              // 6 biases + padded phi bias
#define LDS_BYTES ((size_t)FLT_OFF*2 + (size_t)NFLOATS*4)

// ---------------------------------------------------------------------------
__device__ __forceinline__ f8 wmma_f16(h16 a, h16 b, f8 c) {
  return __builtin_amdgcn_wmma_f32_16x16x32_f16(
      /*neg_a=*/false, a, /*neg_b=*/false, b,
      /*c_mod=*/(short)0, c, /*reuse_a=*/false, /*reuse_b=*/false);
}

// B fragment: 32x16 f16, weights stored [n][k]; lane<16 -> K k0..k0+15,
// lane>=16 -> K k0+16..k0+31 (contiguous k order used here, not the ISA table order), each a contiguous 32B run.
__device__ __forceinline__ h16 loadB(const _Float16* w, int pitch, int n0,
                                     int k0, int lane) {
  const int n = n0 + (lane & 15);
  const int k = k0 + ((lane >> 4) << 4);
  const _Float16* p = w + n * pitch + k;
  h16 b;
  *(h8*)&b       = *(const h8*)p;
  *((h8*)&b + 1) = *(const h8*)(p + 8);
  return b;
}

// A fragment: 16x32 f16 from row-major state strip; lane<16 -> K k0+0..15,
// lane>=16 -> K k0+16..31 (matches loadB).
__device__ __forceinline__ h16 loadA(const _Float16* st, int k0, int lane) {
  const int m  = lane & 15;
  const int hv = (lane >> 4) << 4;  // 0 or 16
  const _Float16* p = st + m * ST_P + k0 + hv;
  h16 a;
  *(h8*)&a       = *(const h8*)p;
  *((h8*)&a + 1) = *(const h8*)(p + 8);
  return a;
}

// C/D fragment layout: lane<16 holds N=lane rows M=0..7; lane>=16 N=lane-16,
// rows M=8..15. Store a C tile into state strip at column base colb+n0.
__device__ __forceinline__ void storeC(_Float16* st, int colb, int n0, f8 c,
                                       int lane) {
  const int n  = colb + n0 + (lane & 15);
  const int mb = (lane >> 4) << 3;
#pragma unroll
  for (int j = 0; j < 8; ++j) st[(mb + j) * ST_P + n] = (_Float16)c[j];
}

__device__ __forceinline__ f8 biasC(const float* b, int n0, int lane) {
  const float v = b[n0 + (lane & 15)];
  f8 c;
#pragma unroll
  for (int j = 0; j < 8; ++j) c[j] = v;
  return c;
}

__device__ __forceinline__ float frcp(float x) {
#if __has_builtin(__builtin_amdgcn_rcpf)
  return __builtin_amdgcn_rcpf(x);        // v_rcp_f32
#else
  return 1.f / x;
#endif
}
__device__ __forceinline__ float ftanh(float x) {
#if __has_builtin(__builtin_amdgcn_tanhf)
  return __builtin_amdgcn_tanhf(x);       // v_tanh_f32 (native on gfx1250)
#else
  return 1.f - 2.f * frcp(__expf(2.f * x) + 1.f);
#endif
}
__device__ __forceinline__ f8 tanh8(f8 v) {
  f8 r;
#pragma unroll
  for (int j = 0; j < 8; ++j) r[j] = ftanh(v[j]);
  return r;
}
__device__ __forceinline__ f8 sigm8(f8 v) {
  f8 r;
#pragma unroll
  for (int j = 0; j < 8; ++j) r[j] = frcp(1.f + __expf(-v[j]));
  return r;
}

// ---------------------------------------------------------------------------
__global__ __launch_bounds__(128, 1) void anima_scan_wmma(
    const float* __restrict__ x,
    const float* __restrict__ W_enc_w, const float* __restrict__ W_enc_b,
    const float* __restrict__ WfW, const float* __restrict__ WfI,
    const float* __restrict__ WfA,
    const float* __restrict__ Wg_w, const float* __restrict__ Wg_b,
    const float* __restrict__ Iz_w, const float* __restrict__ Iz_b,
    const float* __restrict__ Ir_w, const float* __restrict__ Ir_b,
    const float* __restrict__ Ih_w, const float* __restrict__ Ih_b,
    const float* __restrict__ AfW, const float* __restrict__ AfI,
    const float* __restrict__ AfA,
    const float* __restrict__ Ag_w, const float* __restrict__ Ag_b,
    const float* __restrict__ phi_w, const float* __restrict__ phi_b,
    float* __restrict__ out) {
  extern __shared__ _Float16 smem[];
  const int tid = threadIdx.x, nthr = blockDim.x;

  // ---- Stage weights (f32 global -> f16 LDS, transposed [n][k]) -----------
  for (int i = tid; i < 64 * 32; i += nthr) {            // encoder, K 8 -> 32
    const int n = i >> 5, k = i & 31;
    smem[ENC_OFF + n * ENC_P + k] =
        (_Float16)(k < SS ? W_enc_w[k * DD + n] : 0.f);
  }
  for (int i = tid; i < 64 * 192; i += nthr) {           // stacked W_in
    const int n = i / 192, k = i - n * 192;
    const float v = (k < 64)    ? WfW[k * DD + n]
                    : (k < 128) ? WfI[(k - 64) * DD + n]
                                : WfA[(k - 128) * DD + n];
    smem[WIN_OFF + n * KP192 + k] = (_Float16)v;
  }
  for (int i = tid; i < 64 * 192; i += nthr) {           // stacked A_in
    const int n = i / 192, k = i - n * 192;
    const float v = (k < 64)    ? AfW[k * DD + n]
                    : (k < 128) ? AfI[(k - 64) * DD + n]
                                : AfA[(k - 128) * DD + n];
    smem[AIN_OFF + n * KP192 + k] = (_Float16)v;
  }
  for (int i = tid; i < 64 * 192; i += nthr) {           // z/r/h weights
    const int n = i / 192, k = i - n * 192;
    smem[IZ_OFF + n * KP192 + k] = (_Float16)Iz_w[k * DD + n];
    smem[IR_OFF + n * KP192 + k] = (_Float16)Ir_w[k * DD + n];
    smem[IH_OFF + n * KP192 + k] = (_Float16)Ih_w[k * DD + n];
  }
  for (int i = tid; i < 64 * 128; i += nthr) {           // gates
    const int n = i >> 7, k = i & 127;
    smem[WG_OFF + n * KP128 + k] = (_Float16)Wg_w[k * DD + n];
    smem[AG_OFF + n * KP128 + k] = (_Float16)Ag_w[k * DD + n];
  }
  for (int i = tid; i < 16 * 64; i += nthr) {            // phi, N 4 -> 16
    const int n = i >> 6, k = i & 63;
    smem[PHI_OFF + n * PHI_P + k] =
        (_Float16)(n < OO ? phi_w[k * OO + n] : 0.f);
  }
  // Biases as f32.
  float* flds = (float*)(smem + FLT_OFF);
  float* fenc = flds;        float* fwg = flds + 64;  float* fiz = flds + 128;
  float* fir  = flds + 192;  float* fih = flds + 256; float* fag = flds + 320;
  float* fphb = flds + 384;  // 16 entries, padded
  if (tid < 64) {
    fenc[tid] = W_enc_b[tid]; fwg[tid] = Wg_b[tid]; fiz[tid] = Iz_b[tid];
    fir[tid]  = Ir_b[tid];    fih[tid] = Ih_b[tid]; fag[tid] = Ag_b[tid];
  }
  if (tid < 16) fphb[tid] = (tid < OO) ? phi_b[tid] : 0.f;
  // Zero all per-wave state strips (W = I = A = 0 at t=0).
  for (int i = tid; i < 4 * ST_WAVE; i += nthr) smem[ST_OFF + i] = (_Float16)0.f;
  __syncthreads();

  const int lane  = threadIdx.x & 31;
  const int wave  = threadIdx.x >> 5;
  const int bBase = blockIdx.x * 64 + wave * 16;  // 16 batch rows per wave
  _Float16* st = smem + ST_OFF + wave * ST_WAVE;
  const _Float16* sENC = smem + ENC_OFF;
  const _Float16* sWIN = smem + WIN_OFF;
  const _Float16* sWG  = smem + WG_OFF;
  const _Float16* sIZ  = smem + IZ_OFF;
  const _Float16* sIR  = smem + IR_OFF;
  const _Float16* sIH  = smem + IH_OFF;
  const _Float16* sAIN = smem + AIN_OFF;
  const _Float16* sAG  = smem + AG_OFF;
  const _Float16* sPHI = smem + PHI_OFF;

  // f32 I-state in C-fragment registers for the elementwise GRU update.
  f8 Ist[4];
#pragma unroll
  for (int nt = 0; nt < 4; ++nt)
#pragma unroll
    for (int j = 0; j < 8; ++j) Ist[nt][j] = 0.f;

  for (int t = 0; t < TT; ++t) {
    // -- encoder A-fragment straight from global x (K padded 8 -> 32) -------
    h16 xa;
#pragma unroll
    for (int j = 0; j < 16; ++j) xa[j] = (_Float16)0.f;
    if (lane < 16) {
      const float* xp = x + ((size_t)(bBase + lane) * TT + t) * SS;
#pragma unroll
      for (int j = 0; j < SS; ++j) xa[j] = (_Float16)xp[j];
      if (t + 1 < TT) __builtin_prefetch(xp + SS, 0, 1);  // global_prefetch_b8
    }

    // -- W phase: W_new = tanh(tanh(x@Wenc+b) + [W,I,A]@stack) * sig(gate) --
    f8 wn[4];
#pragma unroll
    for (int nt = 0; nt < 4; ++nt) {
      const int n0 = nt * 16;
      f8 c = biasC(fenc, n0, lane);
      c = wmma_f16(xa, loadB(sENC, ENC_P, n0, 0, lane), c);
      c = tanh8(c);  // xt contribution, already in C layout
#pragma unroll
      for (int kc = 0; kc < 6; ++kc)
        c = wmma_f16(loadA(st, kc * 32, lane),
                     loadB(sWIN, KP192, n0, kc * 32, lane), c);
      f8 g = biasC(fwg, n0, lane);
#pragma unroll
      for (int kc = 0; kc < 4; ++kc)  // gate input [I, A] = cols 64..191
        g = wmma_f16(loadA(st, 64 + kc * 32, lane),
                     loadB(sWG, KP128, n0, kc * 32, lane), g);
      wn[nt] = tanh8(c) * sigm8(g);
    }
#pragma unroll
    for (int nt = 0; nt < 4; ++nt) storeC(st, 0, nt * 16, wn[nt], lane);

    // -- z gate over comb = [W_new, I, A] (cols 0..191) ---------------------
    f8 z[4];
#pragma unroll
    for (int nt = 0; nt < 4; ++nt) {
      const int n0 = nt * 16;
      f8 c = biasC(fiz, n0, lane);
#pragma unroll
      for (int kc = 0; kc < 6; ++kc)
        c = wmma_f16(loadA(st, kc * 32, lane),
                     loadB(sIZ, KP192, n0, kc * 32, lane), c);
      z[nt] = sigm8(c);
    }
    // -- r gate, immediately folded into r*I scratch (cols 192..255) --------
#pragma unroll
    for (int nt = 0; nt < 4; ++nt) {
      const int n0 = nt * 16;
      f8 c = biasC(fir, n0, lane);
#pragma unroll
      for (int kc = 0; kc < 6; ++kc)
        c = wmma_f16(loadA(st, kc * 32, lane),
                     loadB(sIR, KP192, n0, kc * 32, lane), c);
      storeC(st, 192, n0, sigm8(c) * Ist[nt], lane);
    }
    // -- h = tanh([W_new, r*I, A] @ I_h); I_new = (1-z)*I + z*h -------------
    const int hcol[6] = {0, 32, 192, 224, 128, 160};
#pragma unroll
    for (int nt = 0; nt < 4; ++nt) {
      const int n0 = nt * 16;
      f8 c = biasC(fih, n0, lane);
#pragma unroll
      for (int kc = 0; kc < 6; ++kc)
        c = wmma_f16(loadA(st, hcol[kc], lane),
                     loadB(sIH, KP192, n0, kc * 32, lane), c);
      const f8 h = tanh8(c);
      f8 In;
#pragma unroll
      for (int j = 0; j < 8; ++j)
        In[j] = (1.f - z[nt][j]) * Ist[nt][j] + z[nt][j] * h[j];
      Ist[nt] = In;
      storeC(st, 64, n0, In, lane);  // old I dead: z/r/rI already consumed it
    }
    // -- A phase: cols 0..191 now hold exactly [W_new, I_new, A_old] --------
    f8 an[4];
#pragma unroll
    for (int nt = 0; nt < 4; ++nt) {
      const int n0 = nt * 16;
      f8 c;
#pragma unroll
      for (int j = 0; j < 8; ++j) c[j] = 0.f;
#pragma unroll
      for (int kc = 0; kc < 6; ++kc)
        c = wmma_f16(loadA(st, kc * 32, lane),
                     loadB(sAIN, KP192, n0, kc * 32, lane), c);
      f8 g = biasC(fag, n0, lane);
#pragma unroll
      for (int kc = 0; kc < 4; ++kc)  // [W_new, I_new] = cols 0..127
        g = wmma_f16(loadA(st, kc * 32, lane),
                     loadB(sAG, KP128, n0, kc * 32, lane), g);
      an[nt] = tanh8(c) * sigm8(g);
    }
#pragma unroll
    for (int nt = 0; nt < 4; ++nt) storeC(st, 128, nt * 16, an[nt], lane);

    // -- action = A_new @ phi + b : padded-N WMMA head (2 more WMMAs) -------
    {
      f8 c = biasC(fphb, 0, lane);
      c = wmma_f16(loadA(st, 128, lane), loadB(sPHI, PHI_P, 0, 0, lane), c);
      c = wmma_f16(loadA(st, 160, lane), loadB(sPHI, PHI_P, 0, 32, lane), c);
      const int n = lane & 15;
      if (n < OO) {
        const int mb = (lane >> 4) << 3;
        float* op = out + ((size_t)(bBase + mb) * TT + t) * OO + n;
#pragma unroll
        for (int j = 0; j < 8; ++j) op[(size_t)j * TT * OO] = c[j];
      }
    }
  }
}

// ---------------------------------------------------------------------------
extern "C" void kernel_launch(void* const* d_in, const int* in_sizes, int n_in,
                              void* d_out, int out_size, void* d_ws,
                              size_t ws_size, hipStream_t stream) {
  (void)in_sizes; (void)n_in; (void)d_ws; (void)ws_size; (void)out_size;
  const float* x        = (const float*)d_in[0];
  const float* W_enc_w  = (const float*)d_in[1];
  const float* W_enc_b  = (const float*)d_in[2];
  const float* WfW      = (const float*)d_in[3];
  const float* WfI      = (const float*)d_in[4];
  const float* WfA      = (const float*)d_in[5];
  const float* Wg_w     = (const float*)d_in[6];
  const float* Wg_b     = (const float*)d_in[7];
  const float* Iz_w     = (const float*)d_in[8];
  const float* Iz_b     = (const float*)d_in[9];
  const float* Ir_w     = (const float*)d_in[10];
  const float* Ir_b     = (const float*)d_in[11];
  const float* Ih_w     = (const float*)d_in[12];
  const float* Ih_b     = (const float*)d_in[13];
  const float* AfW      = (const float*)d_in[14];
  const float* AfI      = (const float*)d_in[15];
  const float* AfA      = (const float*)d_in[16];
  const float* Ag_w     = (const float*)d_in[17];
  const float* Ag_b     = (const float*)d_in[18];
  const float* phi_w    = (const float*)d_in[19];
  const float* phi_b    = (const float*)d_in[20];
  float* out = (float*)d_out;

  // 64 batch tiles of 16 rows; 4 waves (128 threads) per workgroup sharing
  // one LDS weight image (~215 KB, fits CDNA5's 320 KB WGP LDS).
  anima_scan_wmma<<<dim3(BB / 64), dim3(128), LDS_BYTES, stream>>>(
      x, W_enc_w, W_enc_b, WfW, WfI, WfA, Wg_w, Wg_b, Iz_w, Iz_b, Ir_w, Ir_b,
      Ih_w, Ih_b, AfW, AfI, AfA, Ag_w, Ag_b, phi_w, phi_b, out);
}
